// HAGMoE_32684701123013
// MI455X (gfx1250) — matrix operand order repaired and checked by reading, not compiled, not run on hardware
//
#include <hip/hip_runtime.h>

// ---------------------------------------------------------------------------
// HAGMoE on MI455X (gfx1250): routed (gathered) expert evaluation with
// v_wmma_f32_16x16x32_bf16, double-buffered LDS stages, and async
// global->LDS (ASYNCcnt) staging of the bf16 activation operands.
// Weight traffic (252 MB fp32) ~11us @ 23.3 TB/s dominates once only the
// 4 experts of each token's selected group run.
// ---------------------------------------------------------------------------

#define B_ROWS  1024
#define D_DIM   512
#define H_DIM   1024
#define O_DIM   512
#define N_G     6        // META*MACRO groups
#define LDSK    40       // LDS row stride in bf16 elems (80B: 16B-aligned chunks)

typedef __attribute__((ext_vector_type(16))) __bf16 v16bf;
typedef __attribute__((ext_vector_type(8)))  __bf16 v8bf;
typedef __attribute__((ext_vector_type(8)))  float  v8f;
typedef __attribute__((ext_vector_type(4)))  float  f32x4;
typedef __attribute__((ext_vector_type(4)))  unsigned int u32x4;
typedef __attribute__((ext_vector_type(4)))  int    i32x4;

// ---- workspace layout (bytes) ----
#define WS_GCOUNT   40        // int[6]
#define WS_ROWLIST  256       // int[6*1024]
#define WS_PROB     24832     // float[1024*8]
#define WS_XBF      57600     // bf16[1024*512]
#define WS_H1BF     1106176   // bf16[1024*4*1024]
#define WS_HBF      9494784   // bf16[1024*4*1024]

// ---------------------------------------------------------------------------
// Async global->LDS copy (gfx1250 GLOBAL_LOAD_ASYNC_TO_LDS_B128, ASYNCcnt).
// ---------------------------------------------------------------------------
#if defined(__AMDGCN__) && __has_builtin(__builtin_amdgcn_global_load_async_to_lds_b128)
#define USE_ASYNC_LDS 1
typedef __attribute__((address_space(1))) i32x4 gas_i32x4;
typedef __attribute__((address_space(3))) i32x4 las_i32x4;
#else
#define USE_ASYNC_LDS 0
#endif

__device__ __forceinline__ void async_copy_b128(const void* gsrc, void* ldst) {
#if USE_ASYNC_LDS
    __builtin_amdgcn_global_load_async_to_lds_b128(
        (gas_i32x4*)gsrc, (las_i32x4*)ldst, 0, 0);
#else
    *(u32x4*)ldst = *(const u32x4*)gsrc;
#endif
}

__device__ __forceinline__ void wait_async() {
#if USE_ASYNC_LDS
#if __has_builtin(__builtin_amdgcn_s_wait_asynccnt)
    __builtin_amdgcn_s_wait_asynccnt(0);
#else
    asm volatile("s_wait_asynccnt 0x0" ::: "memory");
#endif
#endif
}

// ---------------------------------------------------------------------------
// WMMA fragment loads from LDS (layouts per CDNA5 ISA 7.12.2)
// A 16x32 bf16: lane<16 -> row=lane, K={0..7,16..23}; lane>=16 -> K={8..15,24..31}
// B 32x16 bf16: lane<16 -> col=lane, K={0..7,16..23}; lane>=16 -> col=lane-16, K={8..15,24..31}
// ---------------------------------------------------------------------------
__device__ __forceinline__ v16bf frag_a(const __bf16* As, int wr, int lane) {
    int row = wr * 16 + (lane & 15);
    int kh  = (lane >> 4) << 3;                 // 0 or 8
    const __bf16* p = As + row * LDSK + kh;
    v8bf lo = *(const v8bf*)(p);
    v8bf hi = *(const v8bf*)(p + 16);
    v16bf r;
#pragma unroll
    for (int i = 0; i < 8; ++i) { r[i] = lo[i]; r[i + 8] = hi[i]; }
    return r;
}

__device__ __forceinline__ v16bf frag_b(const __bf16* Bs, int wc, int lane) {
    int col  = wc * 16 + (lane & 15);
    int koff = (lane >> 4) << 3;                // 0 or 8
    const __bf16* p = Bs + col * LDSK + koff;   // Bs stored transposed [n][k]
    v8bf lo = *(const v8bf*)(p);
    v8bf hi = *(const v8bf*)(p + 16);
    v16bf r;
#pragma unroll
    for (int i = 0; i < 8; ++i) { r[i] = lo[i]; r[i + 8] = hi[i]; }
    return r;
}

// A stage: 64 rows x 32 K bf16; one async b128 per thread (addr precomputed).
__device__ __forceinline__ void stage_load_a(__bf16* As, const __bf16* aptr, int t) {
    int row = t >> 2, ch = t & 3;
    async_copy_b128(aptr, As + row * LDSK + ch * 8);
}

// B stage: 32 K x 64 N f32 weights -> bf16, stored transposed Bs[n][k].
// Each thread: 2 consecutive K rows x 4 N -> packed bf16 pairs, 4 b32 stores.
__device__ __forceinline__ void stage_load_b(__bf16* Bs, const float* bptr,
                                             int ldw, int t) {
    int kp = (t >> 4) << 1;   // 0,2,...,30
    int n4 = (t & 15) << 2;   // 0,4,...,60
    f32x4 v0 = *(const f32x4*)bptr;            // row kp
    f32x4 v1 = *(const f32x4*)(bptr + ldw);    // row kp+1
#pragma unroll
    for (int i = 0; i < 4; ++i) {
        union { __bf16 h[2]; unsigned int u; } pk;
        pk.h[0] = (__bf16)v0[i];
        pk.h[1] = (__bf16)v1[i];
        *(unsigned int*)(Bs + (n4 + i) * LDSK + kp) = pk.u;
    }
}

__device__ __forceinline__ void stage_mma(const __bf16* As, const __bf16* Bs,
                                          int wr, int wc2, int lane,
                                          v8f& acc0, v8f& acc1) {
    v16bf a  = frag_a(As, wr, lane);
    v16bf b0 = frag_b(Bs, wc2 * 2 + 0, lane);
    v16bf b1 = frag_b(Bs, wc2 * 2 + 1, lane);
    acc0 = __builtin_amdgcn_wmma_f32_16x16x32_bf16(false, a, false, b0,
                                                   (short)0, acc0, false, false);
    acc1 = __builtin_amdgcn_wmma_f32_16x16x32_bf16(false, a, false, b1,
                                                   (short)0, acc1, false, false);
}

// One full double-buffered K sweep (nk chunks of 32) accumulating into acc0/1.
// Leaves the block barrier-synchronized on exit (buffers safe to reuse).
__device__ __forceinline__ void gemm_phase(const __bf16* aptr, const float* bptr,
                                           int ldw, int nk,
                                           __bf16* As0, __bf16* As1,
                                           __bf16* Bs0, __bf16* Bs1,
                                           int t, int wr, int wc2, int lane,
                                           v8f& acc0, v8f& acc1) {
    __bf16* Asb[2] = {As0, As1};
    __bf16* Bsb[2] = {Bs0, Bs1};
    stage_load_a(Asb[0], aptr, t);
    stage_load_b(Bsb[0], bptr, ldw, t);
    wait_async();
    __syncthreads();
    for (int s = 1; s <= nk; ++s) {
        if (s < nk) {
            const float* bn = bptr + (size_t)s * 32 * ldw;
#if defined(__AMDGCN__)
            __builtin_prefetch(bn + (size_t)32 * ldw, 0, 3);   // near hint, 1 stage ahead
#endif
            stage_load_a(Asb[s & 1], aptr + s * 32, t);
            stage_load_b(Bsb[s & 1], bn, ldw, t);
        }
        stage_mma(Asb[(s - 1) & 1], Bsb[(s - 1) & 1], wr, wc2, lane, acc0, acc1);
        wait_async();
        __syncthreads();
    }
}

// ---------------------------------------------------------------------------
// K0: zero output + workspace counters (ws is not re-poisoned between replays)
// ---------------------------------------------------------------------------
__global__ __launch_bounds__(256) void zero_kernel(float* out, float* wsf) {
    int i = blockIdx.x * 256 + threadIdx.x;
    if (i < B_ROWS * O_DIM + 1) out[i] = 0.0f;
    if (i < 64) wsf[i] = 0.0f;       // covers gcount[6] too (bit pattern 0)
}

// ---------------------------------------------------------------------------
// K1: router.  One block per token: 8 dot products over D=512, argmax routing,
// per-group row list, softmax prob record for aux loss, x -> bf16 conversion.
// ---------------------------------------------------------------------------
__global__ __launch_bounds__(256) void router_kernel(
    const float* __restrict__ x, const float* __restrict__ meta_w,
    const float* __restrict__ meta_b, const float* __restrict__ macro_w,
    const float* __restrict__ macro_b, __bf16* __restrict__ xbf,
    float* __restrict__ prob, int* __restrict__ gcount, int* __restrict__ rowlist)
{
    const int b = blockIdx.x;
    const int t = threadIdx.x;
    const float* xr = x + (size_t)b * D_DIM;
    float p[8];
#pragma unroll
    for (int v = 0; v < 8; ++v) p[v] = 0.0f;
    for (int i = t; i < D_DIM; i += 256) {
        float xv = xr[i];
        xbf[(size_t)b * D_DIM + i] = (__bf16)xv;
        p[0] += xv * meta_w[i * 2 + 0];
        p[1] += xv * meta_w[i * 2 + 1];
#pragma unroll
        for (int m = 0; m < 2; ++m)
#pragma unroll
            for (int k = 0; k < 3; ++k)
                p[2 + m * 3 + k] += xv * macro_w[(size_t)(m * D_DIM + i) * 3 + k];
    }
    __shared__ float red[8][256];
#pragma unroll
    for (int v = 0; v < 8; ++v) red[v][t] = p[v];
    __syncthreads();
    for (int s = 128; s > 0; s >>= 1) {
        if (t < s)
#pragma unroll
            for (int v = 0; v < 8; ++v) red[v][t] += red[v][t + s];
        __syncthreads();
    }
    if (t == 0) {
        float ml0 = red[0][0] + meta_b[0];
        float ml1 = red[1][0] + meta_b[1];
        int   mi  = (ml1 > ml0) ? 1 : 0;            // first-max tie-break
        float mmx = ml0 > ml1 ? ml0 : ml1;
        float e0 = expf(ml0 - mmx), e1 = expf(ml1 - mmx);
        float inv = 1.0f / (e0 + e1);
        float l[3];
#pragma unroll
        for (int k = 0; k < 3; ++k) l[k] = red[2 + mi * 3 + k][0] + macro_b[mi * 3 + k];
        int ki = 0;
        if (l[1] > l[ki]) ki = 1;
        if (l[2] > l[ki]) ki = 2;
        float lmx = l[0];
        if (l[1] > lmx) lmx = l[1];
        if (l[2] > lmx) lmx = l[2];
        float q0 = expf(l[0] - lmx), q1 = expf(l[1] - lmx), q2 = expf(l[2] - lmx);
        float qi = 1.0f / (q0 + q1 + q2);
        int grp = mi * 3 + ki;
        int pos = atomicAdd(&gcount[grp], 1);
        rowlist[grp * B_ROWS + pos] = b;
        float* pr = prob + b * 8;
        pr[0] = e0 * inv; pr[1] = e1 * inv; pr[2] = (float)mi;
        pr[3] = q0 * qi;  pr[4] = q1 * qi;  pr[5] = q2 * qi;
    }
}

// ---------------------------------------------------------------------------
// K2: h1 = relu(x @ fc1[e] + b1) -> bf16, gathered rows of the expert's group.
// grid: (H/64, B/64, 24); block 256 = 8 waves, each owns a 16x32 C strip.
// ---------------------------------------------------------------------------
__global__ __launch_bounds__(256) void fc1_kernel(
    const __bf16* __restrict__ xbf, const float* __restrict__ fc1_w,
    const float* __restrict__ fc1_b, __bf16* __restrict__ h1bf,
    const int* __restrict__ gcount, const int* __restrict__ rowlist)
{
    const int g = blockIdx.z >> 2;
    const int j = blockIdx.z & 3;
    const int e = blockIdx.z;
    const int cnt = gcount[g];
    const int row0 = blockIdx.y * 64;
    if (row0 >= cnt) return;
    const int n0 = blockIdx.x * 64;
    __shared__ __align__(16) __bf16 As0[64 * LDSK], As1[64 * LDSK];
    __shared__ __align__(16) __bf16 Bs0[64 * LDSK], Bs1[64 * LDSK];
    __shared__ int rid[64];
    const int t = threadIdx.x;
    if (t < 64) {
        int s  = row0 + t;
        int sc = s < cnt ? s : cnt - 1;
        rid[t] = rowlist[g * B_ROWS + sc];
    }
    __syncthreads();
    const int lane = t & 31, w = t >> 5, wr = w & 3, wc2 = w >> 2;
    const int arow = t >> 2, ach = t & 3;
    const int bk2 = (t >> 4) << 1, bn4 = (t & 15) << 2;
    const __bf16* aptr = xbf + (size_t)rid[arow] * D_DIM + ach * 8;
    const float*  bptr = fc1_w + (size_t)e * D_DIM * H_DIM
                       + (size_t)bk2 * H_DIM + n0 + bn4;
    v8f acc0 = {}, acc1 = {};
    gemm_phase(aptr, bptr, H_DIM, D_DIM / 32, As0, As1, Bs0, Bs1,
               t, wr, wc2, lane, acc0, acc1);
    const int col0 = n0 + wc2 * 32 + (lane & 15);
    const float bias0 = fc1_b[e * H_DIM + col0];
    const float bias1 = fc1_b[e * H_DIM + col0 + 16];
    const int half = (lane >> 4) << 3;
#pragma unroll
    for (int i = 0; i < 8; ++i) {
        int ml = wr * 16 + half + i;
        if (row0 + ml < cnt) {
            size_t rb = ((size_t)rid[ml] * 4 + j) * H_DIM;
            float v0 = acc0[i] + bias0;
            float v1 = acc1[i] + bias1;
            h1bf[rb + col0]      = (__bf16)(v0 > 0.0f ? v0 : 0.0f);
            h1bf[rb + col0 + 16] = (__bf16)(v1 > 0.0f ? v1 : 0.0f);
        }
    }
}

// ---------------------------------------------------------------------------
// K3: h = relu(h1 @ fc2[e] + x @ proj[e] + b2 + bp) -> bf16 (two fused phases)
// ---------------------------------------------------------------------------
__global__ __launch_bounds__(256) void fc2_kernel(
    const __bf16* __restrict__ h1bf, const __bf16* __restrict__ xbf,
    const float* __restrict__ fc2_w, const float* __restrict__ fc2_b,
    const float* __restrict__ proj_w, const float* __restrict__ proj_b,
    __bf16* __restrict__ hbf, const int* __restrict__ gcount,
    const int* __restrict__ rowlist)
{
    const int g = blockIdx.z >> 2;
    const int j = blockIdx.z & 3;
    const int e = blockIdx.z;
    const int cnt = gcount[g];
    const int row0 = blockIdx.y * 64;
    if (row0 >= cnt) return;
    const int n0 = blockIdx.x * 64;
    __shared__ __align__(16) __bf16 As0[64 * LDSK], As1[64 * LDSK];
    __shared__ __align__(16) __bf16 Bs0[64 * LDSK], Bs1[64 * LDSK];
    __shared__ int rid[64];
    const int t = threadIdx.x;
    if (t < 64) {
        int s  = row0 + t;
        int sc = s < cnt ? s : cnt - 1;
        rid[t] = rowlist[g * B_ROWS + sc];
    }
    __syncthreads();
    const int lane = t & 31, w = t >> 5, wr = w & 3, wc2 = w >> 2;
    const int arow = t >> 2, ach = t & 3;
    const int bk2 = (t >> 4) << 1, bn4 = (t & 15) << 2;
    const int r_a = rid[arow];
    v8f acc0 = {}, acc1 = {};
    // phase 1: h1 @ fc2
    {
        const __bf16* aptr = h1bf + ((size_t)r_a * 4 + j) * H_DIM + ach * 8;
        const float*  bptr = fc2_w + (size_t)e * H_DIM * H_DIM
                           + (size_t)bk2 * H_DIM + n0 + bn4;
        gemm_phase(aptr, bptr, H_DIM, H_DIM / 32, As0, As1, Bs0, Bs1,
                   t, wr, wc2, lane, acc0, acc1);
    }
    // phase 2: x @ proj
    {
        const __bf16* aptr = xbf + (size_t)r_a * D_DIM + ach * 8;
        const float*  bptr = proj_w + (size_t)e * D_DIM * H_DIM
                           + (size_t)bk2 * H_DIM + n0 + bn4;
        gemm_phase(aptr, bptr, H_DIM, D_DIM / 32, As0, As1, Bs0, Bs1,
                   t, wr, wc2, lane, acc0, acc1);
    }
    const int col0 = n0 + wc2 * 32 + (lane & 15);
    const float bias0 = fc2_b[e * H_DIM + col0] + proj_b[e * H_DIM + col0];
    const float bias1 = fc2_b[e * H_DIM + col0 + 16] + proj_b[e * H_DIM + col0 + 16];
    const int half = (lane >> 4) << 3;
#pragma unroll
    for (int i = 0; i < 8; ++i) {
        int ml = wr * 16 + half + i;
        if (row0 + ml < cnt) {
            size_t rb = ((size_t)rid[ml] * 4 + j) * H_DIM;
            float v0 = acc0[i] + bias0;
            float v1 = acc1[i] + bias1;
            hbf[rb + col0]      = (__bf16)(v0 > 0.0f ? v0 : 0.0f);
            hbf[rb + col0 + 16] = (__bf16)(v1 > 0.0f ? v1 : 0.0f);
        }
    }
}

// ---------------------------------------------------------------------------
// K4: final = 0.25 * sum_{j=0..3} (h_j @ fc3[g*4+j] + b3).  Experts looped
// inside the block in fixed order -> no atomics, deterministic output.
// ---------------------------------------------------------------------------
__global__ __launch_bounds__(256) void fc3_kernel(
    const __bf16* __restrict__ hbf, const float* __restrict__ fc3_w,
    const float* __restrict__ fc3_b, float* __restrict__ out,
    const int* __restrict__ gcount, const int* __restrict__ rowlist)
{
    const int g = blockIdx.z;
    const int cnt = gcount[g];
    const int row0 = blockIdx.y * 64;
    if (row0 >= cnt) return;
    const int n0 = blockIdx.x * 64;
    __shared__ __align__(16) __bf16 As0[64 * LDSK], As1[64 * LDSK];
    __shared__ __align__(16) __bf16 Bs0[64 * LDSK], Bs1[64 * LDSK];
    __shared__ int rid[64];
    const int t = threadIdx.x;
    if (t < 64) {
        int s  = row0 + t;
        int sc = s < cnt ? s : cnt - 1;
        rid[t] = rowlist[g * B_ROWS + sc];
    }
    __syncthreads();
    const int lane = t & 31, w = t >> 5, wr = w & 3, wc2 = w >> 2;
    const int arow = t >> 2, ach = t & 3;
    const int bk2 = (t >> 4) << 1, bn4 = (t & 15) << 2;
    const size_t abase = (size_t)rid[arow] * 4 * H_DIM + ach * 8;
    v8f acc0 = {}, acc1 = {};
    for (int j = 0; j < 4; ++j) {
        const __bf16* aptr = hbf + abase + (size_t)j * H_DIM;
        const float*  bptr = fc3_w + (size_t)(g * 4 + j) * H_DIM * O_DIM
                           + (size_t)bk2 * O_DIM + n0 + bn4;
        gemm_phase(aptr, bptr, O_DIM, H_DIM / 32, As0, As1, Bs0, Bs1,
                   t, wr, wc2, lane, acc0, acc1);
    }
    const int col0 = n0 + wc2 * 32 + (lane & 15);
    float bsum0 = 0.0f, bsum1 = 0.0f;
#pragma unroll
    for (int j = 0; j < 4; ++j) {
        bsum0 += fc3_b[(g * 4 + j) * O_DIM + col0];
        bsum1 += fc3_b[(g * 4 + j) * O_DIM + col0 + 16];
    }
    const int half = (lane >> 4) << 3;
#pragma unroll
    for (int i = 0; i < 8; ++i) {
        int ml = wr * 16 + half + i;
        if (row0 + ml < cnt) {
            size_t rb = (size_t)rid[ml] * O_DIM;
            out[rb + col0]      = 0.25f * (acc0[i] + bsum0);
            out[rb + col0 + 16] = 0.25f * (acc1[i] + bsum1);
        }
    }
}

// ---------------------------------------------------------------------------
// K5: aux load-balancing loss, deterministic serial reduction over 1024 rows.
// ---------------------------------------------------------------------------
__global__ void finalize_aux(const float* __restrict__ prob, float* __restrict__ out) {
    if (threadIdx.x != 0 || blockIdx.x != 0) return;
    float mp0 = 0.0f, mp1 = 0.0f;
    float ps[6] = {0, 0, 0, 0, 0, 0};
    float cn[2] = {0, 0};
    for (int b = 0; b < B_ROWS; ++b) {
        const float* pr = prob + b * 8;
        mp0 += pr[0];
        mp1 += pr[1];
        int m = (int)pr[2];
        cn[m] += 1.0f;
        ps[m * 3 + 0] += pr[3];
        ps[m * 3 + 1] += pr[4];
        ps[m * 3 + 2] += pr[5];
    }
    mp0 *= (1.0f / B_ROWS);
    mp1 *= (1.0f / B_ROWS);
    float aux = 0.01f * 2.0f * (mp0 * mp0 + mp1 * mp1);
    for (int m = 0; m < 2; ++m) {
        if (cn[m] > 0.0f) {
            float s = 0.0f;
            for (int k = 0; k < 3; ++k) {
                float q = ps[m * 3 + k] / cn[m];
                s += q * q;
            }
            aux += 0.01f * 3.0f * s;
        }
    }
    out[B_ROWS * O_DIM] = aux;
}

// ---------------------------------------------------------------------------
extern "C" void kernel_launch(void* const* d_in, const int* in_sizes, int n_in,
                              void* d_out, int out_size, void* d_ws, size_t ws_size,
                              hipStream_t stream)
{
    const float* x       = (const float*)d_in[0];
    const float* meta_w  = (const float*)d_in[1];
    const float* meta_b  = (const float*)d_in[2];
    const float* macro_w = (const float*)d_in[3];
    const float* macro_b = (const float*)d_in[4];
    const float* fc1_w   = (const float*)d_in[5];
    const float* fc1_b   = (const float*)d_in[6];
    const float* fc2_w   = (const float*)d_in[7];
    const float* fc2_b   = (const float*)d_in[8];
    const float* fc3_w   = (const float*)d_in[9];
    const float* fc3_b   = (const float*)d_in[10];
    const float* proj_w  = (const float*)d_in[11];
    const float* proj_b  = (const float*)d_in[12];
    float* out = (float*)d_out;

    char* ws = (char*)d_ws;
    float*  wsf     = (float*)ws;
    int*    gcount  = (int*)(ws + WS_GCOUNT);
    int*    rowlist = (int*)(ws + WS_ROWLIST);
    float*  prob    = (float*)(ws + WS_PROB);
    __bf16* xbf     = (__bf16*)(ws + WS_XBF);
    __bf16* h1bf    = (__bf16*)(ws + WS_H1BF);
    __bf16* hbf     = (__bf16*)(ws + WS_HBF);

    zero_kernel<<<(B_ROWS * O_DIM + 1 + 255) / 256, 256, 0, stream>>>(out, wsf);
    router_kernel<<<B_ROWS, 256, 0, stream>>>(x, meta_w, meta_b, macro_w, macro_b,
                                              xbf, prob, gcount, rowlist);
    fc1_kernel<<<dim3(H_DIM / 64, B_ROWS / 64, 24), 256, 0, stream>>>(
        xbf, fc1_w, fc1_b, h1bf, gcount, rowlist);
    fc2_kernel<<<dim3(H_DIM / 64, B_ROWS / 64, 24), 256, 0, stream>>>(
        h1bf, xbf, fc2_w, fc2_b, proj_w, proj_b, hbf, gcount, rowlist);
    fc3_kernel<<<dim3(O_DIM / 64, B_ROWS / 64, N_G), 256, 0, stream>>>(
        hbf, fc3_w, fc3_b, out, gcount, rowlist);
    finalize_aux<<<1, 64, 0, stream>>>(prob, out);
}
